// theTransformer_16552803959209
// MI455X (gfx1250) — matrix operand order repaired and checked by reading, not compiled, not run on hardware
//
#include <hip/hip_runtime.h>
#include <hip/hip_bf16.h>
#include <math.h>

// ---------------------------------------------------------------------------
// MI455X (gfx1250) transformer forward. Compute-bound (~675 GFLOP, all data
// resident in the 192MB L2), so: (1) every GEMM-like op goes through
// v_wmma_f32_16x16x32_bf16, (2) weights are converted to bf16 (pre-transposed
// to [N][K]) once per launch, (3) activations are produced in bf16 by the
// producing kernel's epilogue, (4) GEMM tiles are staged into LDS with the
// gfx1250 async global->LDS path (double-buffered) when the toolchain
// exposes the builtins, overlapping loads with WMMA issue.
// ---------------------------------------------------------------------------

#define BSZ    4
#define TSEQ   1024
#define DKC    1024
#define NHEAD  16
#define DH     64
#define DFF    4096
#define NVOCAB 65
#define NLAYER 6

typedef __bf16 bf16_t;
typedef bf16_t v16bf __attribute__((ext_vector_type(16)));
typedef float  v8f   __attribute__((ext_vector_type(8)));

union FragBF { v16bf v; uint4 q[2]; unsigned short s[16]; };

// ---- async global->LDS support (gfx1250 ASYNCcnt path), guarded ----------
#if defined(__has_builtin)
#if __has_builtin(__builtin_amdgcn_global_load_async_to_lds_b128) && \
    __has_builtin(__builtin_amdgcn_s_wait_asynccnt)
#define USE_ASYNC_LDS 1
#endif
#endif
#ifndef USE_ASYNC_LDS
#define USE_ASYNC_LDS 0
#endif

#if USE_ASYNC_LDS
// Builtin params are pointers to 16B int vectors in AS1 (global) / AS3 (LDS).
typedef int v4i_t __attribute__((vector_size(4 * sizeof(int))));
typedef __attribute__((address_space(1))) v4i_t gv4i_t;
typedef __attribute__((address_space(3))) v4i_t lv4i_t;
__device__ __forceinline__ void async_cp16(const void* g, void* l) {
  __builtin_amdgcn_global_load_async_to_lds_b128((gv4i_t*)g, (lv4i_t*)l, 0, 0);
}
#endif

__device__ __forceinline__ unsigned short f2bf(float f) {
  union { __bf16 b; unsigned short u; } cv;
  cv.b = (__bf16)f;                       // let the HW cvt op do the rounding
  return cv.u;
}

__device__ __forceinline__ void store4bf(__bf16* p, float4 f) {
  union { __bf16 b[4]; uint2 u; } cv;
  cv.b[0] = (__bf16)f.x; cv.b[1] = (__bf16)f.y;
  cv.b[2] = (__bf16)f.z; cv.b[3] = (__bf16)f.w;
  *(uint2*)p = cv.u;
}

// ---------------------------------------------------------------------------
// One-time (per launch) weight conversion: W[K][N] f32 -> WT[N][K] bf16.
// Weights are shared across all 6 layers, so this amortizes 6x.
// ---------------------------------------------------------------------------
__global__ __launch_bounds__(256)
void convert_wT(const float* __restrict__ W, __bf16* __restrict__ WT,
                int K, int N) {
  int idx = blockIdx.x * 256 + threadIdx.x;
  if (idx >= K * N) return;
  int k = idx / N, n = idx - k * N;
  WT[(size_t)n * K + k] = (__bf16)W[idx];
}

// ---------------------------------------------------------------------------
// Embedding: x = Wtok[tok] + Wpos[t]  (f32 + bf16 mirror)
// ---------------------------------------------------------------------------
__global__ __launch_bounds__(256)
void embed_kernel(const int* __restrict__ tok, const float* __restrict__ Wtok,
                  const float* __restrict__ Wpos, float* __restrict__ x,
                  __bf16* __restrict__ xbf) {
  int bt = blockIdx.x;
  int t  = bt & (TSEQ - 1);
  int c  = threadIdx.x * 4;
  int id = tok[bt];
  float4 a = *(const float4*)(Wtok + (size_t)id * DKC + c);
  float4 p = *(const float4*)(Wpos + (size_t)t  * DKC + c);
  a.x += p.x; a.y += p.y; a.z += p.z; a.w += p.w;
  *(float4*)(x + (size_t)bt * DKC + c) = a;
  store4bf(xbf + (size_t)bt * DKC + c, a);
}

// ---------------------------------------------------------------------------
// bf16 WMMA GEMM: C[M,N] = A[M,K] @ WT[N,K]^T + bias.
// OUT: 0 = f32 out, 1 = bf16 out, 2 = bf16 out + ReLU.  GUARD: N-guarded.
// Block 256 thr (8 waves), tile 128x128x32; each wave 64x32 (8 wmma accums).
// Both LDS tiles are contiguous-along-K -> pure 16B copies (async capable).
// ---------------------------------------------------------------------------
__device__ __forceinline__ void gemm_step(const unsigned short* __restrict__ tA,
                                          const unsigned short* __restrict__ tB,
                                          int wr0, int wc0, int ln16, int sel,
                                          v8f (&acc)[4][2]) {
  constexpr int LD = 40;
  FragBF af[4], bfr[2];
#pragma unroll
  for (int rt = 0; rt < 4; ++rt) {
    const unsigned short* p = &tA[(wr0 + rt * 16 + ln16) * LD + sel * 8];
    af[rt].q[0] = *(const uint4*)p;
    af[rt].q[1] = *(const uint4*)(p + 16);
  }
#pragma unroll
  for (int ct = 0; ct < 2; ++ct) {
    const unsigned short* p = &tB[(wc0 + ct * 16 + ln16) * LD + sel * 8];
    bfr[ct].q[0] = *(const uint4*)p;
    bfr[ct].q[1] = *(const uint4*)(p + 16);
  }
#pragma unroll
  for (int rt = 0; rt < 4; ++rt)
#pragma unroll
    for (int ct = 0; ct < 2; ++ct)
      acc[rt][ct] = __builtin_amdgcn_wmma_f32_16x16x32_bf16(
          false, af[rt].v, false, bfr[ct].v, (short)0, acc[rt][ct],
          false, false);
}

__device__ __forceinline__ void stage_sync(unsigned short* dA,
                                           unsigned short* dB,
                                           const __bf16* A, const __bf16* WT,
                                           int m0, int n0, int k0, int K,
                                           int N, int tid, bool guard) {
  constexpr int LD = 40;
#pragma unroll
  for (int it = 0; it < 2; ++it) {
    int c = it * 256 + tid, row = c >> 2, part = c & 3;
    *(uint4*)&dA[row * LD + part * 8] =
        *(const uint4*)(A + (size_t)(m0 + row) * K + k0 + part * 8);
    uint4 wv; wv.x = wv.y = wv.z = wv.w = 0u;
    if (!guard || (n0 + row) < N)
      wv = *(const uint4*)(WT + (size_t)(n0 + row) * K + k0 + part * 8);
    *(uint4*)&dB[row * LD + part * 8] = wv;
  }
}

#if USE_ASYNC_LDS
__device__ __forceinline__ void stage_async(unsigned short* dA,
                                            unsigned short* dB,
                                            const __bf16* A, const __bf16* WT,
                                            int m0, int n0, int k0, int K,
                                            int tid) {
  constexpr int LD = 40;
#pragma unroll
  for (int it = 0; it < 2; ++it) {
    int c = it * 256 + tid, row = c >> 2, part = c & 3;
    async_cp16(A  + (size_t)(m0 + row) * K + k0 + part * 8,
               &dA[row * LD + part * 8]);
    async_cp16(WT + (size_t)(n0 + row) * K + k0 + part * 8,
               &dB[row * LD + part * 8]);
  }
}
#endif

template<int OUT, int GUARD>
__global__ __launch_bounds__(256)
void gemm_bf(const __bf16* __restrict__ A, const __bf16* __restrict__ WT,
             const float* __restrict__ bias, void* __restrict__ Cout,
             int M, int N, int K) {
  constexpr int BM = 128, BK = 32, LD = BK + 8;
#if USE_ASYNC_LDS
  constexpr int NBUF = 2;
#else
  constexpr int NBUF = 1;
#endif
  __shared__ __align__(16) unsigned short lA[NBUF][BM * LD];
  __shared__ __align__(16) unsigned short lB[NBUF][BM * LD];

  const int tid  = threadIdx.x;
  const int lane = tid & 31;
  const int wid  = tid >> 5;
  const int ln16 = lane & 15;
  const int sel  = lane >> 4;
  const int wr0  = (wid >> 2) * 64;
  const int wc0  = (wid & 3) * 32;
  const int m0   = blockIdx.y * BM;
  const int n0   = blockIdx.x * BM;
  const int nk   = K / BK;

  v8f zero = {};
  v8f acc[4][2];
#pragma unroll
  for (int i = 0; i < 4; ++i)
#pragma unroll
    for (int j = 0; j < 2; ++j) acc[i][j] = zero;

#if USE_ASYNC_LDS
  if (GUARD == 0) {
    // double-buffered: async-load next tile while WMMAs chew the current one
    stage_async(lA[0], lB[0], A, WT, m0, n0, 0, K, tid);
    for (int kt = 0; kt < nk; ++kt) {
      __builtin_amdgcn_s_wait_asynccnt(0);
      __syncthreads();
      if (kt + 1 < nk)
        stage_async(lA[(kt + 1) & 1], lB[(kt + 1) & 1], A, WT, m0, n0,
                    (kt + 1) * BK, K, tid);
      gemm_step(lA[kt & 1], lB[kt & 1], wr0, wc0, ln16, sel, acc);
    }
  } else
#endif
  {
    for (int kt = 0; kt < nk; ++kt) {
      stage_sync(lA[0], lB[0], A, WT, m0, n0, kt * BK, K, N, tid, GUARD != 0);
      __syncthreads();
      gemm_step(lA[0], lB[0], wr0, wc0, ln16, sel, acc);
      __syncthreads();
    }
  }

  // epilogue: bias (+relu), C/D layout: M = j + 8*sel, N = ln16
#pragma unroll
  for (int rt = 0; rt < 4; ++rt)
#pragma unroll
    for (int ct = 0; ct < 2; ++ct)
#pragma unroll
      for (int j = 0; j < 8; ++j) {
        int m = m0 + wr0 + rt * 16 + j + 8 * sel;
        int n = n0 + wc0 + ct * 16 + ln16;
        if (!GUARD || n < N) {
          float c = acc[rt][ct][j] + bias[n];
          if (OUT == 2) c = fmaxf(c, 0.0f);
          if (OUT == 0) ((float*)Cout)[(size_t)m * N + n] = c;
          else          ((__bf16*)Cout)[(size_t)m * N + n] = (__bf16)c;
        }
      }
}

// ---------------------------------------------------------------------------
// Flash attention (causal), one wave per (b, h, 16-row q tile), bf16 q/k/v.
// ---------------------------------------------------------------------------
__global__ __launch_bounds__(32)
void attention_kernel(const __bf16* __restrict__ Q, const __bf16* __restrict__ K,
                      const __bf16* __restrict__ V, float* __restrict__ O) {
  constexpr int LD = 40;
  __shared__ __align__(16) unsigned short lP[16 * LD];  // [qrow][key_local]
  __shared__ __align__(16) unsigned short lV[64 * LD];  // [dh][key_local]

  const int lane = threadIdx.x & 31;
  const int ln16 = lane & 15;
  const int sel  = lane >> 4;
  const int q0   = blockIdx.x * 16;
  const int hh   = blockIdx.y;
  const int bb   = blockIdx.z;
  const float scl = 0.03125f;             // 1/sqrt(1024)

  // Q A-fragments: memory layout == fragment layout, raw 16B loads
  FragBF qa[2];
  {
    const __bf16* qrow = Q + ((size_t)(bb * TSEQ) + q0 + ln16) * DKC + hh * DH;
#pragma unroll
    for (int fz = 0; fz < 2; ++fz) {
      qa[fz].q[0] = *(const uint4*)(qrow + fz * 32 + sel * 8);
      qa[fz].q[1] = *(const uint4*)(qrow + fz * 32 + 16 + sel * 8);
    }
  }

  v8f zero = {};
  v8f oacc[4];
#pragma unroll
  for (int ct = 0; ct < 4; ++ct) oacc[ct] = zero;
  float mrun[8], lrun[8];
#pragma unroll
  for (int j = 0; j < 8; ++j) { mrun[j] = -__builtin_inff(); lrun[j] = 0.0f; }

  const int nkb = (q0 + 16 + 31) >> 5;
  for (int blkk = 0; blkk < nkb; ++blkk) {
    const int kb = blkk * 32;

    // S = Q K^T for 32 keys (2 tiles x 2 wmma); K frags are raw 16B loads
    v8f slo = zero, shi = zero;
#pragma unroll
    for (int kh = 0; kh < 2; ++kh) {
      const __bf16* krow =
          K + ((size_t)(bb * TSEQ) + kb + kh * 16 + ln16) * DKC + hh * DH +
          sel * 8;
      FragBF kf0, kf1;
      kf0.q[0] = *(const uint4*)(krow);      kf0.q[1] = *(const uint4*)(krow + 16);
      kf1.q[0] = *(const uint4*)(krow + 32); kf1.q[1] = *(const uint4*)(krow + 48);
      v8f t = (kh == 0) ? slo : shi;
      t = __builtin_amdgcn_wmma_f32_16x16x32_bf16(false, qa[0].v, false, kf0.v,
                                                  (short)0, t, false, false);
      t = __builtin_amdgcn_wmma_f32_16x16x32_bf16(false, qa[1].v, false, kf1.v,
                                                  (short)0, t, false, false);
      if (kh == 0) slo = t; else shi = t;
    }

    // stage V block transposed: lV[d][key] (bf16 repack, no conversion)
    {
      const __bf16* vrow = V + ((size_t)(bb * TSEQ) + kb + lane) * DKC + hh * DH;
#pragma unroll
      for (int i = 0; i < 8; ++i) {
        union { uint4 u; unsigned short s[8]; } cv;
        cv.u = *(const uint4*)(vrow + i * 8);
#pragma unroll
        for (int e = 0; e < 8; ++e) lV[(i * 8 + e) * LD + lane] = cv.s[e];
      }
    }

    // online softmax per row (row = q0 + j + 8*sel, key = kb + ln16 [+16])
    const bool needMask = (kb + 31 > q0);
#pragma unroll
    for (int j = 0; j < 8; ++j) {
      int row = q0 + j + 8 * sel;
      float s0 = slo[j] * scl;
      float s1 = shi[j] * scl;
      if (needMask) {
        if (kb + ln16 > row)      s0 = -__builtin_inff();
        if (kb + 16 + ln16 > row) s1 = -__builtin_inff();
      }
      float mx = fmaxf(s0, s1);
      mx = fmaxf(mx, __shfl_xor(mx, 1));
      mx = fmaxf(mx, __shfl_xor(mx, 2));
      mx = fmaxf(mx, __shfl_xor(mx, 4));
      mx = fmaxf(mx, __shfl_xor(mx, 8));
      float mnew = fmaxf(mrun[j], mx);
      float p0 = __expf(s0 - mnew);
      float p1 = __expf(s1 - mnew);
      float rs = p0 + p1;
      rs += __shfl_xor(rs, 1);
      rs += __shfl_xor(rs, 2);
      rs += __shfl_xor(rs, 4);
      rs += __shfl_xor(rs, 8);
      float corr = __expf(mrun[j] - mnew);
      lrun[j] = lrun[j] * corr + rs;
      mrun[j] = mnew;
#pragma unroll
      for (int ct = 0; ct < 4; ++ct) oacc[ct][j] *= corr;
      int rl = j + 8 * sel;
      lP[rl * LD + ln16]      = f2bf(p0);
      lP[rl * LD + 16 + ln16] = f2bf(p1);
    }
    __syncthreads();

    // O += P @ V
    FragBF pf;
    {
      const unsigned short* pp = &lP[ln16 * LD + sel * 8];
      pf.q[0] = *(const uint4*)pp;
      pf.q[1] = *(const uint4*)(pp + 16);
    }
#pragma unroll
    for (int ct = 0; ct < 4; ++ct) {
      FragBF vf;
      const unsigned short* vp = &lV[(ct * 16 + ln16) * LD + sel * 8];
      vf.q[0] = *(const uint4*)vp;
      vf.q[1] = *(const uint4*)(vp + 16);
      oacc[ct] = __builtin_amdgcn_wmma_f32_16x16x32_bf16(
          false, pf.v, false, vf.v, (short)0, oacc[ct], false, false);
    }
    __syncthreads();
  }

  // finalize: O /= l, scatter back to [B,T,C] f32
#pragma unroll
  for (int j = 0; j < 8; ++j) {
    float inv = 1.0f / lrun[j];
    int row = q0 + j + 8 * sel;
#pragma unroll
    for (int ct = 0; ct < 4; ++ct)
      O[((size_t)(bb * TSEQ) + row) * DKC + hh * DH + ct * 16 + ln16] =
          oacc[ct][j] * inv;
  }
}

// ---------------------------------------------------------------------------
// In-place x = x + layer_norm(x)*w, plus bf16 mirror for downstream GEMMs.
// ---------------------------------------------------------------------------
__global__ __launch_bounds__(256)
void add_layernorm(float* __restrict__ x, __bf16* __restrict__ xbf,
                   const float* __restrict__ w) {
  __shared__ float redS[8], redQ[8];
  float* xr = x + (size_t)blockIdx.x * DKC;
  const int tid = threadIdx.x, lane = tid & 31, wid = tid >> 5;

  float4 f = *(const float4*)(xr + tid * 4);
  float s  = f.x + f.y + f.z + f.w;
  float s2 = f.x * f.x + f.y * f.y + f.z * f.z + f.w * f.w;
#pragma unroll
  for (int m = 1; m < 32; m <<= 1) {
    s  += __shfl_xor(s, m);
    s2 += __shfl_xor(s2, m);
  }
  if (lane == 0) { redS[wid] = s; redQ[wid] = s2; }
  __syncthreads();
  if (tid == 0) {
    float a = 0.f, b = 0.f;
    for (int i = 0; i < 8; ++i) { a += redS[i]; b += redQ[i]; }
    redS[0] = a; redQ[0] = b;
  }
  __syncthreads();
  float mean = redS[0] * (1.0f / DKC);
  float var  = redQ[0] * (1.0f / DKC) - mean * mean;
  float rstd = rsqrtf(var + 1e-5f);
  float4 wv = *(const float4*)(w + tid * 4);
  float4 y;
  y.x = f.x + (f.x - mean) * rstd * wv.x;
  y.y = f.y + (f.y - mean) * rstd * wv.y;
  y.z = f.z + (f.z - mean) * rstd * wv.z;
  y.w = f.w + (f.w - mean) * rstd * wv.w;
  *(float4*)(xr + tid * 4) = y;
  store4bf(xbf + (size_t)blockIdx.x * DKC + tid * 4, y);
}

// ---------------------------------------------------------------------------
// probs[b,t,v] = softmax over axis=1 (T) of logits. One WG per (b, v).
// ---------------------------------------------------------------------------
__global__ __launch_bounds__(256)
void softmax_T(const float* __restrict__ logits, float* __restrict__ probs) {
  __shared__ float red[8], bcM, bcS;
  int b  = blockIdx.x / NVOCAB;
  int vv = blockIdx.x - b * NVOCAB;
  const float* base = logits + (size_t)b * TSEQ * NVOCAB + vv;
  float* pbase = probs + (size_t)b * TSEQ * NVOCAB + vv;
  const int tid = threadIdx.x, lane = tid & 31, wid = tid >> 5;

  float vals[4];
  float mx = -__builtin_inff();
#pragma unroll
  for (int i = 0; i < 4; ++i) {
    vals[i] = base[(size_t)(tid + i * 256) * NVOCAB];
    mx = fmaxf(mx, vals[i]);
  }
#pragma unroll
  for (int m = 1; m < 32; m <<= 1) mx = fmaxf(mx, __shfl_xor(mx, m));
  if (lane == 0) red[wid] = mx;
  __syncthreads();
  if (tid == 0) {
    float t = red[0];
    for (int i = 1; i < 8; ++i) t = fmaxf(t, red[i]);
    bcM = t;
  }
  __syncthreads();
  float MX = bcM;
  float e[4], s = 0.f;
#pragma unroll
  for (int i = 0; i < 4; ++i) { e[i] = __expf(vals[i] - MX); s += e[i]; }
#pragma unroll
  for (int m = 1; m < 32; m <<= 1) s += __shfl_xor(s, m);
  if (lane == 0) red[wid] = s;
  __syncthreads();
  if (tid == 0) {
    float t = 0.f;
    for (int i = 0; i < 8; ++i) t += red[i];
    bcS = t;
  }
  __syncthreads();
  float inv = 1.0f / bcS;
#pragma unroll
  for (int i = 0; i < 4; ++i)
    pbase[(size_t)(tid + i * 256) * NVOCAB] = e[i] * inv;
}

// ---------------------------------------------------------------------------
// loss = -mean over B*T rows of log_softmax_V(probs)[target]. Single block.
// ---------------------------------------------------------------------------
__global__ __launch_bounds__(256)
void loss_kernel(const float* __restrict__ probs, const int* __restrict__ tgt,
                 float* __restrict__ out0) {
  __shared__ float red[8];
  const int tid = threadIdx.x, lane = tid & 31, wid = tid >> 5;
  float acc = 0.f;
  for (int r = tid; r < BSZ * TSEQ; r += 256) {
    const float* row = probs + (size_t)r * NVOCAB;
    float mx = -__builtin_inff();
    for (int j = 0; j < NVOCAB; ++j) mx = fmaxf(mx, row[j]);
    float s = 0.f;
    for (int j = 0; j < NVOCAB; ++j) s += __expf(row[j] - mx);
    int tg = tgt[r];
    acc -= (row[tg] - mx) - __logf(s);
  }
#pragma unroll
  for (int m = 1; m < 32; m <<= 1) acc += __shfl_xor(acc, m);
  if (lane == 0) red[wid] = acc;
  __syncthreads();
  if (tid == 0) {
    float t = 0.f;
    for (int i = 0; i < 8; ++i) t += red[i];
    out0[0] = t * (1.0f / (BSZ * TSEQ));
  }
}

// ---------------------------------------------------------------------------
extern "C" void kernel_launch(void* const* d_in, const int* in_sizes, int n_in,
                              void* d_out, int out_size, void* d_ws,
                              size_t ws_size, hipStream_t stream) {
  (void)in_sizes; (void)n_in; (void)out_size; (void)ws_size;
  const int*   batchData = (const int*)d_in[0];
  const int*   target    = (const int*)d_in[1];
  const float* Wtok = (const float*)d_in[3];
  const float* Wpos = (const float*)d_in[4];
  const float* Wq = (const float*)d_in[5];  const float* bq = (const float*)d_in[6];
  const float* Wk = (const float*)d_in[7];  const float* bk = (const float*)d_in[8];
  const float* Wv = (const float*)d_in[9];  const float* bv = (const float*)d_in[10];
  const float* W1 = (const float*)d_in[11]; const float* b1 = (const float*)d_in[12];
  const float* W2 = (const float*)d_in[13]; const float* b2 = (const float*)d_in[14];
  const float* lnw = (const float*)d_in[15];
  const float* Wf = (const float*)d_in[16]; const float* bfb = (const float*)d_in[17];
  // n_layers/n_heads are reference constants (6 / 16).

  const int M = BSZ * TSEQ;                  // 4096
  const size_t ACT = (size_t)M * DKC;        // 4,194,304 elems

  // ---- workspace carve-up (bytes) ----
  char* wsb = (char*)d_ws;
  size_t off = 0;
  auto takeF = [&](size_t elems) { float* p = (float*)(wsb + off);
                                   off += elems * 4; return p; };
  auto takeB = [&](size_t elems) { __bf16* p = (__bf16*)(wsb + off);
                                   off += elems * 2; return p; };
  float*  xf    = takeF(ACT);
  float*  hf    = takeF(ACT);
  __bf16* xbf   = takeB(ACT);
  __bf16* hbf   = takeB(ACT);
  __bf16* qbf   = takeB(ACT);
  __bf16* kbf   = takeB(ACT);
  __bf16* vbf   = takeB(ACT);
  (void)takeB(ACT);                          // pad so ffh spans 4*ACT
  __bf16* ffhbf = qbf;                       // [M, DFF] reuses q/k/v/pad
  float*  logits = takeF((size_t)M * NVOCAB);
  __bf16* WqT = takeB((size_t)DKC * DKC);
  __bf16* WkT = takeB((size_t)DKC * DKC);
  __bf16* WvT = takeB((size_t)DKC * DKC);
  __bf16* W1T = takeB((size_t)DKC * DFF);
  __bf16* W2T = takeB((size_t)DFF * DKC);
  __bf16* WfT = takeB((size_t)DKC * NVOCAB);

  float* out   = (float*)d_out;
  float* probs = out + 1;

  dim3 blk(256);
  dim3 ggQKV(DKC / 128, M / 128);            // (8, 32)
  dim3 ggFF1(DFF / 128, M / 128);            // (32, 32)
  dim3 ggLOG(1, M / 128);
  dim3 agrid(TSEQ / 16, NHEAD, BSZ);

  // one-time bf16 transposed weight mirrors (weights shared by all layers)
  auto cvt = [&](const float* W, __bf16* WT, int K, int N) {
    convert_wT<<<dim3((K * N + 255) / 256), blk, 0, stream>>>(W, WT, K, N);
  };
  cvt(Wq, WqT, DKC, DKC);
  cvt(Wk, WkT, DKC, DKC);
  cvt(Wv, WvT, DKC, DKC);
  cvt(W1, W1T, DKC, DFF);
  cvt(W2, W2T, DFF, DKC);
  cvt(Wf, WfT, DKC, NVOCAB);

  embed_kernel<<<dim3(M), blk, 0, stream>>>(batchData, Wtok, Wpos, xf, xbf);

  for (int layer = 0; layer < NLAYER; ++layer) {
    gemm_bf<1, 0><<<ggQKV, blk, 0, stream>>>(xbf, WqT, bq, qbf, M, DKC, DKC);
    gemm_bf<1, 0><<<ggQKV, blk, 0, stream>>>(xbf, WkT, bk, kbf, M, DKC, DKC);
    gemm_bf<1, 0><<<ggQKV, blk, 0, stream>>>(xbf, WvT, bv, vbf, M, DKC, DKC);
    attention_kernel<<<agrid, dim3(32), 0, stream>>>(qbf, kbf, vbf, hf);
    add_layernorm<<<dim3(M), blk, 0, stream>>>(hf, hbf, lnw);
    gemm_bf<2, 0><<<ggFF1, blk, 0, stream>>>(hbf, W1T, b1, ffhbf, M, DFF, DKC);
    gemm_bf<0, 0><<<ggQKV, blk, 0, stream>>>(ffhbf, W2T, b2, xf, M, DKC, DFF);
    add_layernorm<<<dim3(M), blk, 0, stream>>>(xf, xbf, lnw);
  }

  gemm_bf<0, 1><<<ggLOG, blk, 0, stream>>>(xbf, WfT, bfb, logits, M, NVOCAB, DKC);
  softmax_T<<<dim3(BSZ * NVOCAB), blk, 0, stream>>>(logits, probs);
  loss_kernel<<<dim3(1), blk, 0, stream>>>(probs, target, out);
}
